// CrossSpeakerAttention_41704132444721
// MI455X (gfx1250) — matrix operand order repaired and checked by reading, not compiled, not run on hardware
//
#include <hip/hip_runtime.h>
#include <hip/hip_bf16.h>
#include <stdint.h>

// ---------------------------------------------------------------------------
// CrossSpeakerAttention for MI455X (gfx1250), wave32 + v_wmma_f32_16x16x32_bf16
//   B=8, T=1024, PD=512, H=8, D=64
// ---------------------------------------------------------------------------

typedef __attribute__((ext_vector_type(16))) __bf16 v16bf;
typedef __attribute__((ext_vector_type(8)))  float  v8f;

union Frag {
    v16bf  v;
    uint4  q[2];
    __bf16 bh[16];
};

// native conversion: lowers to v_cvt_pk_bf16_f32
__device__ __forceinline__ uint32_t pack2bf(float lo, float hi) {
    union { uint32_t u; __bf16 b[2]; } p;
    p.b[0] = (__bf16)lo;
    p.b[1] = (__bf16)hi;
    return p.u;
}

#define NB 8
#define TT 1024
#define PDIM 512
#define HH 8
#define DD 64
#define NEG_INF (-__builtin_inff())

// ===========================================================================
// Kernel 0: one-time fp32 -> bf16 conversion of the 4 weight matrices.
// ===========================================================================
__global__ __launch_bounds__(256)
void wcvt_kernel(const float* __restrict__ Wq, const float* __restrict__ Wk,
                 const float* __restrict__ Wv, const float* __restrict__ Wo,
                 __bf16* __restrict__ Wall)
{
    const int tg = blockIdx.x * 256 + threadIdx.x;     // 0..131071
    const int m  = tg >> 15;                           // matrix 0..3
    const int e0 = (tg & 32767) * 8;
    const float* src = (m == 0) ? Wq : (m == 1) ? Wk : (m == 2) ? Wv : Wo;
    const float4 f0 = *(const float4*)(src + e0);
    const float4 f1 = *(const float4*)(src + e0 + 4);
    uint4 pk;
    pk.x = pack2bf(f0.x, f0.y);
    pk.y = pack2bf(f0.z, f0.w);
    pk.z = pack2bf(f1.x, f1.y);
    pk.w = pack2bf(f1.z, f1.w);
    *(uint4*)(Wall + (size_t)m * PDIM * PDIM + e0) = pk;
}

// ===========================================================================
// Kernel 1: fused Q/K/V projection.  out = X @ W^T, X fp32 [8192,512].
// Q,K stored bf16 [b][h][t][d]; V stored transposed bf16 [b][h][d][t].
// grid = 512 blocks (one 16-row m-tile), 256 threads (8 waves).
// Each wave: 4 n-tiles per matrix (4 register accumulators); one LDS
// A-fragment load feeds 4 WMMAs.  mat loop fully unrolled (uniform stores).
// ===========================================================================
__global__ __launch_bounds__(256, 1)
void qkv_proj_kernel(const float* __restrict__ X,
                     const __bf16* __restrict__ Wqb,
                     const __bf16* __restrict__ Wkb,
                     const __bf16* __restrict__ Wvb,
                     __bf16* __restrict__ Qb,
                     __bf16* __restrict__ Kb,
                     __bf16* __restrict__ Vt)
{
    __shared__ __bf16 xs[16 * PDIM];     // 16 KB bf16 tile of X

    const int m0  = blockIdx.x * 16;
    const int tid = threadIdx.x;

    // cooperative load + fp32->bf16 convert of the 16x512 X tile
    {
        const int row = tid >> 4;
        const int col = (tid & 15) * 32;
        const float* src = X + (size_t)(m0 + row) * PDIM + col;
        #pragma unroll
        for (int c = 0; c < 32; c += 8) {
            const float4 f0 = *(const float4*)(src + c);
            const float4 f1 = *(const float4*)(src + c + 4);
            uint4 pk;
            pk.x = pack2bf(f0.x, f0.y);
            pk.y = pack2bf(f0.z, f0.w);
            pk.z = pack2bf(f1.x, f1.y);
            pk.w = pack2bf(f1.z, f1.w);
            *(uint4*)(xs + row * PDIM + col + c) = pk;   // ds_store_b128
        }
    }
    __syncthreads();

    const int wv = tid >> 5;
    const int ln = tid & 31;
    const int lhalf = ln >> 4;     // 0 or 1
    const int lq    = ln & 15;
    const int b  = m0 >> 10;
    const int t0 = m0 & (TT - 1);
    const int ntbase = wv * 4;     // this wave's 4 n-tiles (per matrix)

    const __bf16* arow = xs + lq * PDIM + lhalf * 8;

    #pragma unroll
    for (int mat = 0; mat < 3; ++mat) {
        const __bf16* W = (mat == 0) ? Wqb : (mat == 1) ? Wkb : Wvb;

        const __bf16* wr0[4];
        #pragma unroll
        for (int c = 0; c < 4; ++c)
            wr0[c] = W + (size_t)((ntbase + c) * 16 + lq) * PDIM + lhalf * 8;

        v8f acc[4];
        #pragma unroll
        for (int c = 0; c < 4; ++c)
            #pragma unroll
            for (int e = 0; e < 8; ++e) acc[c][e] = 0.f;

        #pragma unroll 2
        for (int kk = 0; kk < 16; ++kk) {
            Frag a;                                   // one LDS A-frag, 4x reuse
            a.q[0] = *(const uint4*)(arow + kk * 32);
            a.q[1] = *(const uint4*)(arow + kk * 32 + 16);
            #pragma unroll
            for (int c = 0; c < 4; ++c) {
                Frag bf;
                bf.q[0] = *(const uint4*)(wr0[c] + kk * 32);
                bf.q[1] = *(const uint4*)(wr0[c] + kk * 32 + 16);
                acc[c] = __builtin_amdgcn_wmma_f32_16x16x32_bf16(
                    false, a.v, false, bf.v, (short)0, acc[c], false, false);
            }
        }

        #pragma unroll
        for (int c = 0; c < 4; ++c) {
            const int n0 = (ntbase + c) * 16;
            const int h  = n0 >> 6;
            const int d0 = n0 & (DD - 1);
            if (mat < 2) {
                // Q/K: [b][h][t][d]; lane owns column d, 8 strided rows (t)
                __bf16* dst = (mat == 0) ? Qb : Kb;
                const int d = d0 + lq;
                #pragma unroll
                for (int r = 0; r < 8; ++r) {
                    const int t = t0 + r + lhalf * 8;
                    dst[(((size_t)b * HH + h) * TT + t) * DD + d] = (__bf16)acc[c][r];
                }
            } else {
                // V transposed: [b][h][d][t]; 8 consecutive t -> one 16B store
                const int d  = d0 + lq;
                const int ts = t0 + lhalf * 8;
                uint4 pk;
                pk.x = pack2bf(acc[c][0], acc[c][1]);
                pk.y = pack2bf(acc[c][2], acc[c][3]);
                pk.z = pack2bf(acc[c][4], acc[c][5]);
                pk.w = pack2bf(acc[c][6], acc[c][7]);
                *(uint4*)(Vt + (((size_t)b * HH + h) * DD + d) * TT + ts) = pk;
            }
        }
    }
}

// ===========================================================================
// Kernel 2: register-resident flash attention, transposed formulation.
//   S^T = K . Q^T  (softmax over j becomes a per-lane column reduction)
//   O^T = V^T . P^T (P^T built in-register via one shfl_xor(16))
// 1 wave = one 16-query tile of one (b,h).  grid 512 x 256 = 4096 waves.
// Causal structure: only key blocks jb <= i0+14 are visited.
// ===========================================================================
__global__ __launch_bounds__(256)
void attn_kernel(const __bf16* __restrict__ Qb,
                 const __bf16* __restrict__ Kb,
                 const __bf16* __restrict__ Vt,
                 const int* __restrict__ spk,
                 const uint8_t* __restrict__ valid,
                 __bf16* __restrict__ Ob)
{
    const int tid = threadIdx.x;
    const int wv  = tid >> 5;
    const int ln  = tid & 31;
    const int task = blockIdx.x * 8 + wv;        // 0..4095
    const int bh = task >> 6;
    const int i0 = (task & 63) * 16;
    const int b  = bh >> 3;
    const int h  = bh & 7;
    const int lhalf = ln >> 4;
    const int lq    = ln & 15;
    const int i     = i0 + lq;                   // this lane's query row

    const size_t bhQ = ((size_t)b * HH + h) * TT * DD;   // [t][d] base
    const size_t bhV = ((size_t)b * HH + h) * DD * TT;   // [d][t] base

    // Q^T B-fragments (two 32-wide d windows), loaded once
    Frag qf[2];
    #pragma unroll
    for (int w = 0; w < 2; ++w) {
        const __bf16* qr = Qb + bhQ + (size_t)(i0 + lq) * DD + w * 32 + lhalf * 8;
        qf[w].q[0] = *(const uint4*)(qr);
        qf[w].q[1] = *(const uint4*)(qr + 16);
    }

    const int spk_i = spk[b * TT + i];

    v8f oacc[4];
    #pragma unroll
    for (int db = 0; db < 4; ++db)
        #pragma unroll
        for (int e = 0; e < 8; ++e) oacc[db][e] = 0.f;

    float m_prev = NEG_INF;
    float lsum   = 0.f;

    const int nsteps = (i0 + 15 + 31) >> 5;      // key blocks of 32 (causal)
    for (int s = 0; s < nsteps; ++s) {
        const int jb = s * 32;

        if (s + 1 < nsteps) {   // prefetch next key block (global_prefetch_b8)
            __builtin_prefetch(Kb + bhQ + (size_t)(jb + 32 + lq) * DD, 0, 1);
            __builtin_prefetch(Vt + bhV + (size_t)lq * TT + jb + 32, 0, 1);
        }

        // ---- scores: two S^T tiles (j rows, i cols), K-dim = d = 64 ----
        v8f st[2];
        #pragma unroll
        for (int t = 0; t < 2; ++t) {
            v8f c = {0.f, 0.f, 0.f, 0.f, 0.f, 0.f, 0.f, 0.f};
            const __bf16* kr = Kb + bhQ + (size_t)(jb + t * 16 + lq) * DD;
            #pragma unroll
            for (int w = 0; w < 2; ++w) {
                Frag a;
                const __bf16* ap = kr + w * 32 + lhalf * 8;
                a.q[0] = *(const uint4*)(ap);
                a.q[1] = *(const uint4*)(ap + 16);
                c = __builtin_amdgcn_wmma_f32_16x16x32_bf16(
                    false, a.v, false, qf[w].v, (short)0, c, false, false);
            }
            st[t] = c;
        }

        // ---- scale + mask (causal | same-speaker | invalid key) ----
        float p[2][8];
        float mblk = NEG_INF;
        #pragma unroll
        for (int t = 0; t < 2; ++t) {
            const int jbl = jb + t * 16 + lhalf * 8;
            const int4  s0 = *(const int4*)(spk + b * TT + jbl);
            const int4  s1 = *(const int4*)(spk + b * TT + jbl + 4);
            const uint2 vm = *(const uint2*)(valid + b * TT + jbl);
            const int sj[8] = {s0.x, s0.y, s0.z, s0.w, s1.x, s1.y, s1.z, s1.w};
            #pragma unroll
            for (int r = 0; r < 8; ++r) {
                const int j = jbl + r;
                const uint32_t vb = ((r < 4 ? vm.x : vm.y) >> ((r & 3) * 8)) & 0xFFu;
                const bool masked = (j >= i) | (sj[r] == spk_i) | (vb == 0u);
                const float sv = masked ? NEG_INF : st[t][r] * 0.125f;
                p[t][r] = sv;
                mblk = fmaxf(mblk, sv);
            }
        }
        mblk = fmaxf(mblk, __shfl_xor(mblk, 16, 32));

        // ---- online softmax (branch-free -inf guards) ----
        const float m_new = fmaxf(m_prev, mblk);
        const float m_use = (m_new == NEG_INF) ? 0.f : m_new;
        const float alpha = __expf(m_prev - m_use);   // 0 when m_prev==-inf
        float rsum = 0.f;
        #pragma unroll
        for (int t = 0; t < 2; ++t)
            #pragma unroll
            for (int r = 0; r < 8; ++r) {
                const float e = __expf(p[t][r] - m_use);  // exp(-inf)=0 if masked
                p[t][r] = e;
                rsum += e;
            }
        rsum += __shfl_xor(rsum, 16, 32);
        lsum = lsum * alpha + rsum;
        m_prev = m_new;

        #pragma unroll
        for (int db = 0; db < 4; ++db)
            #pragma unroll
            for (int e = 0; e < 8; ++e) oacc[db][e] *= alpha;

        // ---- build P^T B-fragment in-register: one shfl_xor(16) ----
        Frag pf;
        #pragma unroll
        for (int r = 0; r < 8; ++r) {
            const float own   = (lhalf == 0) ? p[1][r] : p[0][r];
            const float cross = __shfl_xor(own, 16, 32);
            pf.bh[r]     = (__bf16)((lhalf == 0) ? p[0][r] : cross);
            pf.bh[r + 8] = (__bf16)((lhalf == 0) ? cross : p[1][r]);
        }

        // ---- O^T += V^T * P^T  (4 d-blocks of 16) ----
        #pragma unroll
        for (int db = 0; db < 4; ++db) {
            Frag a;
            const __bf16* ap =
                Vt + bhV + (size_t)(db * 16 + lq) * TT + jb + lhalf * 16;
            a.q[0] = *(const uint4*)(ap);
            a.q[1] = *(const uint4*)(ap + 8);
            oacc[db] = __builtin_amdgcn_wmma_f32_16x16x32_bf16(
                false, a.v, false, pf.v, (short)0, oacc[db], false, false);
        }
    }

    // ---- normalize + store O as bf16 [b][t][h*64+d] ----
    const float inv = (lsum > 0.f) ? (1.f / lsum) : 0.f;   // all-masked -> 0
    #pragma unroll
    for (int db = 0; db < 4; ++db) {
        uint4 pk;
        pk.x = pack2bf(oacc[db][0] * inv, oacc[db][1] * inv);
        pk.y = pack2bf(oacc[db][2] * inv, oacc[db][3] * inv);
        pk.z = pack2bf(oacc[db][4] * inv, oacc[db][5] * inv);
        pk.w = pack2bf(oacc[db][6] * inv, oacc[db][7] * inv);
        const size_t off = ((size_t)(b * TT + i0 + lq)) * PDIM
                         + h * DD + db * 16 + lhalf * 8;
        *(uint4*)(Ob + off) = pk;
    }
}

// ===========================================================================
// Kernel 3: output projection  Y = O @ Wo^T + bo   (fp32 out)
// grid 512 x 256 -> 4096 waves = 512 m-tiles x 8 n-groups of 4 tiles.
// One global A-fragment load feeds 4 WMMAs.
// ===========================================================================
__global__ __launch_bounds__(256, 1)
void out_proj_kernel(const __bf16* __restrict__ Ob,
                     const __bf16* __restrict__ Wob,
                     const float* __restrict__ bo,
                     float* __restrict__ Y)
{
    const int tid = threadIdx.x;
    const int wv  = tid >> 5;
    const int ln  = tid & 31;
    const int task = blockIdx.x * 8 + wv;        // 0..4095
    const int m0  = (task >> 3) * 16;
    const int ng  = (task & 7) * 4;              // first of 4 n-tiles
    const int lhalf = ln >> 4;
    const int lq    = ln & 15;

    const __bf16* ar0 = Ob + (size_t)(m0 + lq) * PDIM + lhalf * 8;
    const __bf16* wr0[4];
    #pragma unroll
    for (int c = 0; c < 4; ++c)
        wr0[c] = Wob + (size_t)((ng + c) * 16 + lq) * PDIM + lhalf * 8;

    v8f acc[4];
    #pragma unroll
    for (int c = 0; c < 4; ++c)
        #pragma unroll
        for (int e = 0; e < 8; ++e) acc[c][e] = 0.f;

    #pragma unroll 2
    for (int kk = 0; kk < 16; ++kk) {
        Frag a;
        a.q[0] = *(const uint4*)(ar0 + kk * 32);
        a.q[1] = *(const uint4*)(ar0 + kk * 32 + 16);
        #pragma unroll
        for (int c = 0; c < 4; ++c) {
            Frag bf;
            bf.q[0] = *(const uint4*)(wr0[c] + kk * 32);
            bf.q[1] = *(const uint4*)(wr0[c] + kk * 32 + 16);
            acc[c] = __builtin_amdgcn_wmma_f32_16x16x32_bf16(
                false, a.v, false, bf.v, (short)0, acc[c], false, false);
        }
    }

    #pragma unroll
    for (int c = 0; c < 4; ++c) {
        const int n = (ng + c) * 16 + lq;
        const float bn = bo[n];
        #pragma unroll
        for (int r = 0; r < 8; ++r) {
            const int m = m0 + r + lhalf * 8;
            Y[(size_t)m * PDIM + n] = acc[c][r] + bn;
        }
    }
}

// ===========================================================================
extern "C" void kernel_launch(void* const* d_in, const int* in_sizes, int n_in,
                              void* d_out, int out_size, void* d_ws, size_t ws_size,
                              hipStream_t stream)
{
    (void)in_sizes; (void)n_in; (void)out_size; (void)ws_size;
    const float*   X     = (const float*)d_in[0];
    const int*     spk   = (const int*)d_in[1];
    const uint8_t* valid = (const uint8_t*)d_in[2];   // jnp bool -> 1 byte/elem
    const float*   Wq    = (const float*)d_in[3];
    const float*   Wk    = (const float*)d_in[4];
    const float*   Wv    = (const float*)d_in[5];
    const float*   Wo    = (const float*)d_in[6];
    const float*   bo    = (const float*)d_in[7];
    float*         out   = (float*)d_out;

    char* ws = (char*)d_ws;
    const size_t SZ = (size_t)NB * HH * TT * DD * sizeof(__bf16);     // 8 MB
    __bf16* Qb   = (__bf16*)(ws);
    __bf16* Kb   = (__bf16*)(ws + SZ);
    __bf16* Vt   = (__bf16*)(ws + 2 * SZ);
    __bf16* Ob   = (__bf16*)(ws + 3 * SZ);
    __bf16* Wall = (__bf16*)(ws + 4 * SZ);            // Wq,Wk,Wv,Wo bf16
    __bf16* Wqb  = Wall;
    __bf16* Wkb  = Wall + PDIM * PDIM;
    __bf16* Wvb  = Wall + 2 * PDIM * PDIM;
    __bf16* Wob  = Wall + 3 * PDIM * PDIM;

    wcvt_kernel    <<<dim3(512), dim3(256), 0, stream>>>(Wq, Wk, Wv, Wo, Wall);
    qkv_proj_kernel<<<dim3(512), dim3(256), 0, stream>>>(X, Wqb, Wkb, Wvb, Qb, Kb, Vt);
    attn_kernel    <<<dim3(512), dim3(256), 0, stream>>>(Qb, Kb, Vt, spk, valid, Ob);
    out_proj_kernel<<<dim3(512), dim3(256), 0, stream>>>(Ob, Wob, bo, out);
}
